// SimpleGNN_43138651521645
// MI455X (gfx1250) — matrix operand order repaired and checked by reading, not compiled, not run on hardware
//
#include <hip/hip_runtime.h>
#include <hip/hip_bf16.h>

// ---------------------------------------------------------------------------
// GAT GNN for MI455X (gfx1250, wave32, WMMA).
// GEMMs (67 GFLOP total) run through v_wmma_f32_16x16x32_f16 with f32 accum.
// Block tile 64Mx128N, 8 waves, each wave = 32x32 accum = 4 WMMA per K-step.
// B tile staged via global_load_async_to_lds_b128 (ASYNCcnt path).
// Edge pipeline (reduces to self-loop attention) is streaming VALU + atomics.
// ---------------------------------------------------------------------------

#ifndef TRY_ASYNC_LDS
#define TRY_ASYNC_LDS 1   // flip to 0 if the async-to-LDS asm is rejected
#endif

typedef _Float16 v16h __attribute__((ext_vector_type(16)));
typedef _Float16 v8h  __attribute__((ext_vector_type(8)));
typedef float    v8f  __attribute__((ext_vector_type(8)));

// ---- ordered-uint encoding so float max can use hardware atomicMax(u32) ----
__device__ __forceinline__ unsigned fenc(float f) {
  unsigned u = __float_as_uint(f);
  return (u & 0x80000000u) ? ~u : (u | 0x80000000u);
}
__device__ __forceinline__ float fdec(unsigned u) {
  return __uint_as_float((u & 0x80000000u) ? (u & 0x7FFFFFFFu) : ~u);
}

// ---------------------------------------------------------------------------
// init helpers
// ---------------------------------------------------------------------------
__global__ void k_zero(float* __restrict__ p, int n) {
  int i = blockIdx.x * blockDim.x + threadIdx.x;
  if (i < n) p[i] = 0.0f;
}
__global__ void k_init_emax(unsigned* __restrict__ e) {
  if (threadIdx.x < 8) e[threadIdx.x] = fenc(-3.4e38f);
}

// convert weights [h][k][o] (f32) -> transposed f16 [h*O+o][k] for B fragments
__global__ void k_cvtT(const float* __restrict__ src, _Float16* __restrict__ dst,
                       int H, int K, int O) {
  int idx = blockIdx.x * blockDim.x + threadIdx.x;
  int total = H * K * O;
  if (idx >= total) return;
  int o = idx % O;
  int k = (idx / O) % K;
  int h = idx / (O * K);
  dst[((size_t)(h * O + o)) * K + k] = (_Float16)src[idx];
}

// ---------------------------------------------------------------------------
// Tiled WMMA GEMM: C[M,N] = opt_relu(A[M,K] * B[K,N] + bias)
// A: f32 row-major (converted to f16 in LDS).  BT: f16, [N][K] (pre-transposed).
// Block = 256 threads (8 waves); tile = 64(M) x 128(N), K step = 32.
// Wave w: wm = w&1 (32-row half), wn = w>>1 (32-col quarter) -> 2x2 WMMA tiles.
// Fragment lane layouts used here (contiguous k order, same for A and B):
//   A (16x32 f16): lane L -> M=L&15, K={kb..kb+15}, kb=(L>>4)*16
//   B (32x16 f16): lane L -> N=L&15, K={kb..kb+15}, kb=(L>>4)*16  (stored [n][k])
//   C/D (16x16 f32): vgpr r -> M = r + 8*(L>>4), N = L&15
// ---------------------------------------------------------------------------
__global__ __launch_bounds__(256) void k_gemm(
    const float* __restrict__ A, const _Float16* __restrict__ BT,
    const float* __restrict__ bias, float* __restrict__ C,
    int M, int K, int N, int relu) {
  __shared__ __align__(16) _Float16 As[64 * 32];   // [m][k]  4 KB
  __shared__ __align__(16) _Float16 Bs[128 * 32];  // [n][k]  8 KB

  const int m0   = blockIdx.x * 64;
  const int n0   = blockIdx.y * 128;
  const int tid  = threadIdx.x;
  const int wave = tid >> 5;
  const int lane = tid & 31;
  const int wm   = wave & 1;   // 0..1 : which 32-row half
  const int wn   = wave >> 1;  // 0..3 : which 32-col quarter

  // A-tile loader: thread converts 8 consecutive f32 -> f16 (one v8h store)
  const int am = tid >> 2;           // 0..63
  const int ak = (tid & 3) * 8;      // 0,8,16,24
  const int amr = (m0 + am < M) ? (m0 + am) : (M - 1);  // clamp (stores guarded)
  const float* Aptr = A + (size_t)amr * K + ak;

  // B-tile loader: thread owns 16 consecutive f16 (32 bytes)
  const int bn = tid >> 1;           // 0..127
  const int bk = (tid & 1) * 16;     // 0,16
  const _Float16* Bptr = BT + (size_t)(n0 + bn) * K + bk;
#if TRY_ASYNC_LDS
  // generic LDS pointer's low 32 bits are the LDS byte address (ISA 10.2)
  const unsigned blds = (unsigned)(size_t)(Bs + bn * 32 + bk);
#endif

  // fragment gather addresses (LDS)
  const int fr  = lane & 15;         // row-in-16 (A) / col-in-16 (B/D)
  const int kbA = (lane >> 4) * 16;
  const int kbB = (lane >> 4) * 16;

  const v8f vzero = {0.f, 0.f, 0.f, 0.f, 0.f, 0.f, 0.f, 0.f};
  v8f acc[2][2] = {{vzero, vzero}, {vzero, vzero}};

  for (int k0 = 0; k0 < K; k0 += 32) {
    // ---- stage A tile: f32 -> f16 through VGPRs (needs conversion) ----
    float4 f0 = *(const float4*)(Aptr + k0);
    float4 f1 = *(const float4*)(Aptr + k0 + 4);
    v8h hv;
    hv[0] = (_Float16)f0.x; hv[1] = (_Float16)f0.y;
    hv[2] = (_Float16)f0.z; hv[3] = (_Float16)f0.w;
    hv[4] = (_Float16)f1.x; hv[5] = (_Float16)f1.y;
    hv[6] = (_Float16)f1.z; hv[7] = (_Float16)f1.w;
    *(v8h*)(As + am * 32 + ak) = hv;

    // ---- stage B tile: raw f16 copy ----
#if TRY_ASYNC_LDS
    {  // async global->LDS, 16B per lane per op, tracked by ASYNCcnt
      unsigned long long ga = (unsigned long long)(size_t)(Bptr + k0);
      asm volatile("global_load_async_to_lds_b128 %0, %1, off"
                   :: "v"(blds), "v"(ga) : "memory");
      asm volatile("global_load_async_to_lds_b128 %0, %1, off offset:16"
                   :: "v"(blds), "v"(ga) : "memory");
    }
#else
    {
      const uint4* bsrc = (const uint4*)(Bptr + k0);
      *(uint4*)(Bs + bn * 32 + bk)     = bsrc[0];
      *(uint4*)(Bs + bn * 32 + bk + 8) = bsrc[1];
    }
#endif

    if (k0 + 32 < K) {  // global_prefetch_b8 of next K-tiles
      __builtin_prefetch((const void*)(Aptr + k0 + 32), 0, 1);
      __builtin_prefetch((const void*)(Bptr + k0 + 32), 0, 1);
    }

#if TRY_ASYNC_LDS
    asm volatile("s_wait_asynccnt 0x0" ::: "memory");
#endif
    __syncthreads();

    // ---- build fragments and issue 4 WMMAs (2 A-frags x 2 B-frags) ----
    v16h af[2], bf[2];
#pragma unroll
    for (int i = 0; i < 2; ++i) {
      int arow = wm * 32 + i * 16 + fr;
      v8h lo = *(const v8h*)(As + arow * 32 + kbA);
      v8h hi = *(const v8h*)(As + arow * 32 + kbA + 8);
      af[i] = __builtin_shufflevector(lo, hi, 0, 1, 2, 3, 4, 5, 6, 7,
                                      8, 9, 10, 11, 12, 13, 14, 15);
    }
#pragma unroll
    for (int j = 0; j < 2; ++j) {
      int bcol = wn * 32 + j * 16 + fr;
      bf[j] = *(const v16h*)(Bs + bcol * 32 + kbB);
    }
#pragma unroll
    for (int i = 0; i < 2; ++i)
#pragma unroll
      for (int j = 0; j < 2; ++j)
        acc[i][j] = __builtin_amdgcn_wmma_f32_16x16x32_f16(
            false, af[i], false, bf[j], (short)0, acc[i][j], false, false);
    __syncthreads();
  }

  // ---- epilogue: D layout -> (row, col) per lane/vgpr ----
  const int rb = (lane >> 4) * 8;
#pragma unroll
  for (int j = 0; j < 2; ++j) {
    const int col = n0 + wn * 32 + j * 16 + fr;
    const float bv = bias ? bias[col] : 0.0f;
#pragma unroll
    for (int i = 0; i < 2; ++i) {
#pragma unroll
      for (int r = 0; r < 8; ++r) {
        int row = m0 + wm * 32 + i * 16 + r + rb;
        if (row < M) {
          float v = acc[i][j][r] + bv;
          if (relu) v = fmaxf(v, 0.0f);
          C[(size_t)row * N + col] = v;
        }
      }
    }
  }
}

// ---------------------------------------------------------------------------
// s_src[n*4+h] = xt[n, h*128 : h*128+128] . a[h][0:128]
// s_dst[n*4+h] = xt[n, ...]              . a[h][128:256]
// One block (4 waves) per node; wave h does head h (wave32 shuffle reduce).
// ---------------------------------------------------------------------------
__global__ __launch_bounds__(128) void k_scores(
    const float* __restrict__ xt, const float* __restrict__ a,
    float* __restrict__ ssrc, float* __restrict__ sdst) {
  const int n = blockIdx.x;
  const int h = threadIdx.x >> 5;
  const int lane = threadIdx.x & 31;
  const float* row  = xt + (size_t)n * 512 + h * 128;
  const float* asrc = a + h * 256;
  const float* adst = asrc + 128;
  float p1 = 0.f, p2 = 0.f;
#pragma unroll
  for (int i = lane; i < 128; i += 32) {
    float v = row[i];
    p1 += v * asrc[i];
    p2 += v * adst[i];
  }
  for (int off = 16; off; off >>= 1) {
    p1 += __shfl_down(p1, off, 32);
    p2 += __shfl_down(p2, off, 32);
  }
  if (lane == 0) {
    ssrc[(size_t)n * 4 + h] = p1;
    sdst[(size_t)n * 4 + h] = p2;
  }
}

// per-head global max of leaky_relu(s_src[src]+s_dst[dst]) over all edges
__global__ __launch_bounds__(256) void k_edge_max(
    const int* __restrict__ src, const int* __restrict__ dst,
    const float* __restrict__ ssrc, const float* __restrict__ sdst,
    unsigned* __restrict__ emax, int E) {
  __shared__ float red[256];
  const int e = blockIdx.x * 256 + threadIdx.x;
  float mh[4] = {-3.4e38f, -3.4e38f, -3.4e38f, -3.4e38f};
  if (e < E) {
    int s = src[e], d = dst[e];
#pragma unroll
    for (int h = 0; h < 4; ++h) {
      float v = ssrc[(size_t)s * 4 + h] + sdst[(size_t)d * 4 + h];
      mh[h] = (v > 0.f) ? v : 0.2f * v;
    }
  }
#pragma unroll
  for (int h = 0; h < 4; ++h) {
    red[threadIdx.x] = mh[h];
    __syncthreads();
    for (int off = 128; off; off >>= 1) {
      if (threadIdx.x < off)
        red[threadIdx.x] = fmaxf(red[threadIdx.x], red[threadIdx.x + off]);
      __syncthreads();
    }
    if (threadIdx.x == 0) atomicMax(emax + h, fenc(red[0]));
    __syncthreads();
  }
}

// denom[dst*4+h] += exp(e - emax[h])
__global__ __launch_bounds__(256) void k_edge_denom(
    const int* __restrict__ src, const int* __restrict__ dst,
    const float* __restrict__ ssrc, const float* __restrict__ sdst,
    const unsigned* __restrict__ emax, float* __restrict__ denom, int E) {
  const int e = blockIdx.x * 256 + threadIdx.x;
  if (e >= E) return;
  int s = src[e], d = dst[e];
#pragma unroll
  for (int h = 0; h < 4; ++h) {
    float v = ssrc[(size_t)s * 4 + h] + sdst[(size_t)d * 4 + h];
    v = (v > 0.f) ? v : 0.2f * v;
    float ex = __expf(v - fdec(emax[h]));
    atomicAdd(denom + (size_t)d * 4 + h, ex);
  }
}

// only self-loop edges survive the reference's diag scatter
__global__ __launch_bounds__(256) void k_edge_self(
    const int* __restrict__ src, const int* __restrict__ dst,
    const float* __restrict__ ssrc, const float* __restrict__ sdst,
    const unsigned* __restrict__ emax, const float* __restrict__ denom,
    float* __restrict__ diag, int E) {
  const int e = blockIdx.x * 256 + threadIdx.x;
  if (e >= E) return;
  int s = src[e], d = dst[e];
  if (s != d) return;
#pragma unroll
  for (int h = 0; h < 4; ++h) {
    float v = ssrc[(size_t)s * 4 + h] + sdst[(size_t)d * 4 + h];
    v = (v > 0.f) ? v : 0.2f * v;
    float ex = __expf(v - fdec(emax[h]));
    diag[(size_t)s * 4 + h] = ex / (denom[(size_t)d * 4 + h] + 1e-16f);
  }
}

// x2 = relu(LN_512(diag[h]*xt1[n,h*128+o]))   (in-place on buf)
__global__ __launch_bounds__(128) void k_h1_ln(
    float* __restrict__ buf, const float* __restrict__ diag,
    const float* __restrict__ g, const float* __restrict__ b) {
  __shared__ float red[128];
  const int n = blockIdx.x, t = threadIdx.x;
  float vals[4];
  float lsum = 0.f;
#pragma unroll
  for (int j = 0; j < 4; ++j) {  // element i = j*128 + t, head = j
    float v = buf[(size_t)n * 512 + j * 128 + t] * diag[(size_t)n * 4 + j];
    vals[j] = v;
    lsum += v;
  }
  red[t] = lsum;
  __syncthreads();
  for (int off = 64; off; off >>= 1) {
    if (t < off) red[t] += red[t + off];
    __syncthreads();
  }
  float mean = red[0] * (1.0f / 512.0f);
  __syncthreads();
  float lvar = 0.f;
#pragma unroll
  for (int j = 0; j < 4; ++j) {
    float d = vals[j] - mean;
    lvar += d * d;
  }
  red[t] = lvar;
  __syncthreads();
  for (int off = 64; off; off >>= 1) {
    if (t < off) red[t] += red[t + off];
    __syncthreads();
  }
  float rstd = rsqrtf(red[0] * (1.0f / 512.0f) + 1e-5f);
#pragma unroll
  for (int j = 0; j < 4; ++j) {
    int i = j * 128 + t;
    float o = (vals[j] - mean) * rstd * g[i] + b[i];
    buf[(size_t)n * 512 + i] = fmaxf(o, 0.0f);
  }
}

// h2 = mean_h(diag2[h]*xt2[h]); out = relu(LN_128(h2))  -> d_out node embeddings
__global__ __launch_bounds__(128) void k_h2_out(
    const float* __restrict__ xt2, const float* __restrict__ diag,
    const float* __restrict__ g, const float* __restrict__ b,
    float* __restrict__ out) {
  __shared__ float red[128];
  const int n = blockIdx.x, t = threadIdx.x;
  float v = 0.f;
#pragma unroll
  for (int h = 0; h < 4; ++h)
    v += diag[(size_t)n * 4 + h] * xt2[(size_t)n * 512 + h * 128 + t];
  v *= 0.25f;
  red[t] = v;
  __syncthreads();
  for (int off = 64; off; off >>= 1) {
    if (t < off) red[t] += red[t + off];
    __syncthreads();
  }
  float mean = red[0] * (1.0f / 128.0f);
  __syncthreads();
  float d = v - mean;
  red[t] = d * d;
  __syncthreads();
  for (int off = 64; off; off >>= 1) {
    if (t < off) red[t] += red[t + off];
    __syncthreads();
  }
  float rstd = rsqrtf(red[0] * (1.0f / 128.0f) + 1e-5f);
  out[(size_t)n * 128 + t] = fmaxf(d * rstd * g[t] + b[t], 0.0f);
}

// per-channel partial sums over 256-node chunks -> atomicAdd to colsum[128]
__global__ __launch_bounds__(128) void k_colsum(
    const float* __restrict__ emb, float* __restrict__ colsum, int M) {
  const int t = threadIdx.x;
  const int n0 = blockIdx.x * 256;
  float s = 0.f;
  for (int i = 0; i < 256; ++i) {
    int n = n0 + i;
    if (n < M) s += emb[(size_t)n * 128 + t];
  }
  atomicAdd(colsum + t, s);
}

// graph_embedding[o] = mean_nodes . pool_W[:,o] + pool_b[o]
__global__ __launch_bounds__(128) void k_pool(
    const float* __restrict__ colsum, const float* __restrict__ W,
    const float* __restrict__ bias, float* __restrict__ out, int M) {
  __shared__ float mean[128];
  const int t = threadIdx.x;
  mean[t] = colsum[t] / (float)M;
  __syncthreads();
  float s = bias[t];
  for (int i = 0; i < 128; ++i) s += mean[i] * W[i * 128 + t];
  out[t] = s;
}

// ---------------------------------------------------------------------------
extern "C" void kernel_launch(void* const* d_in, const int* in_sizes, int n_in,
                              void* d_out, int out_size, void* d_ws, size_t ws_size,
                              hipStream_t stream) {
  const float* nf    = (const float*)d_in[0];
  const int*   ei    = (const int*)d_in[1];
  const float* projW = (const float*)d_in[2];
  const float* projB = (const float*)d_in[3];
  const float* W1    = (const float*)d_in[4];
  const float* a1    = (const float*)d_in[5];
  const float* W2    = (const float*)d_in[6];
  const float* a2    = (const float*)d_in[7];
  const float* ln1g  = (const float*)d_in[8];
  const float* ln1b  = (const float*)d_in[9];
  const float* ln2g  = (const float*)d_in[10];
  const float* ln2b  = (const float*)d_in[11];
  const float* poolW = (const float*)d_in[12];
  const float* poolB = (const float*)d_in[13];

  const int M = in_sizes[0] / 64;   // 100000 nodes
  const int E = in_sizes[1] / 2;    // 3.2M edges
  const int* src = ei;
  const int* dst = ei + E;

  // workspace carve-out (~470 MB)
  char* ws = (char*)d_ws;
  size_t off = 0;
  auto alloc = [&](size_t bytes) -> size_t {
    size_t o = off;
    off += (bytes + 255) & ~(size_t)255;
    return o;
  };
  float*    x    = (float*)(ws + alloc((size_t)M * 128 * 4));
  float*    bufA = (float*)(ws + alloc((size_t)M * 512 * 4));  // xt1 -> x2 in place
  float*    bufB = (float*)(ws + alloc((size_t)M * 512 * 4));  // xt2
  _Float16* BT0  = (_Float16*)(ws + alloc((size_t)128 * 64 * 2));
  _Float16* BT1  = (_Float16*)(ws + alloc((size_t)512 * 128 * 2));
  _Float16* BT2  = (_Float16*)(ws + alloc((size_t)512 * 512 * 2));
  float*    ssrc1 = (float*)(ws + alloc((size_t)M * 4 * 4));
  float*    sdst1 = (float*)(ws + alloc((size_t)M * 4 * 4));
  float*    ssrc2 = (float*)(ws + alloc((size_t)M * 4 * 4));
  float*    sdst2 = (float*)(ws + alloc((size_t)M * 4 * 4));
  const size_t zero_floats = (size_t)M * 16 + 128;  // denoms+diags+colsum
  float* zbase  = (float*)(ws + alloc(zero_floats * 4));
  float* denom1 = zbase;
  float* diag1  = zbase + (size_t)M * 4;
  float* denom2 = zbase + (size_t)M * 8;
  float* diag2  = zbase + (size_t)M * 12;
  float* colsum = zbase + (size_t)M * 16;
  unsigned* emax1 = (unsigned*)(ws + alloc(64));
  unsigned* emax2 = emax1 + 4;

  float* out = (float*)d_out;
  const int gx = (M + 63) / 64;       // 64-row block tiles
  const int eg = (E + 255) / 256;

  // per-call (graph-replay-safe) init
  {
    int zn = (int)zero_floats;
    k_zero<<<(zn + 255) / 256, 256, 0, stream>>>(zbase, zn);
  }
  k_init_emax<<<1, 32, 0, stream>>>(emax1);

  // weight conversion: f32 [h][k][o] -> f16 [h*O+o][k]
  k_cvtT<<<(64 * 128 + 255) / 256, 256, 0, stream>>>(projW, BT0, 1, 64, 128);
  k_cvtT<<<(4 * 128 * 128 + 255) / 256, 256, 0, stream>>>(W1, BT1, 4, 128, 128);
  k_cvtT<<<(4 * 512 * 128 + 255) / 256, 256, 0, stream>>>(W2, BT2, 4, 512, 128);

  // x = relu(nf @ proj_W + b)            [M,64]x[64,128]
  k_gemm<<<dim3(gx, 1), 256, 0, stream>>>(nf, BT0, projB, x, M, 64, 128, 1);
  // xt1 = x @ W1cat                      [M,128]x[128,512]
  k_gemm<<<dim3(gx, 4), 256, 0, stream>>>(x, BT1, nullptr, bufA, M, 128, 512, 0);

  // layer-1 attention (only self-loops contribute to output)
  k_scores<<<M, 128, 0, stream>>>(bufA, a1, ssrc1, sdst1);
  k_edge_max<<<eg, 256, 0, stream>>>(src, dst, ssrc1, sdst1, emax1, E);
  k_edge_denom<<<eg, 256, 0, stream>>>(src, dst, ssrc1, sdst1, emax1, denom1, E);
  k_edge_self<<<eg, 256, 0, stream>>>(src, dst, ssrc1, sdst1, emax1, denom1, diag1, E);
  k_h1_ln<<<M, 128, 0, stream>>>(bufA, diag1, ln1g, ln1b);  // bufA now holds x2

  // xt2 = x2 @ W2cat                     [M,512]x[512,512]
  k_gemm<<<dim3(gx, 4), 256, 0, stream>>>(bufA, BT2, nullptr, bufB, M, 512, 512, 0);

  // layer-2 attention
  k_scores<<<M, 128, 0, stream>>>(bufB, a2, ssrc2, sdst2);
  k_edge_max<<<eg, 256, 0, stream>>>(src, dst, ssrc2, sdst2, emax2, E);
  k_edge_denom<<<eg, 256, 0, stream>>>(src, dst, ssrc2, sdst2, emax2, denom2, E);
  k_edge_self<<<eg, 256, 0, stream>>>(src, dst, ssrc2, sdst2, emax2, denom2, diag2, E);

  // node embeddings -> d_out, then pooled graph embedding
  k_h2_out<<<M, 128, 0, stream>>>(bufB, diag2, ln2g, ln2b, out);
  k_colsum<<<(M + 255) / 256, 128, 0, stream>>>(out, colsum, M);
  k_pool<<<1, 128, 0, stream>>>(colsum, poolW, poolB, out + (size_t)M * 128, M);
}
